// STSSTransformation_22402549416232
// MI455X (gfx1250) — compile-verified, matrix operand order checked by reading, not run on hardware
//
#include <hip/hip_runtime.h>
#include <hip/hip_bf16.h>

typedef __attribute__((ext_vector_type(16))) _Float16 v16h;
typedef __attribute__((ext_vector_type(8)))  _Float16 v8h;
typedef __attribute__((ext_vector_type(4)))  _Float16 v4h;
typedef __attribute__((ext_vector_type(8)))  float    v8f;

#define BT    32
#define DIN   256
#define DHID  64
#define HIN   112
#define WINP  112
#define HO    56
#define WO    56
#define NPIX  (BT*HO*WO)          // 100352
#define YN_HALFS ((size_t)NPIX * DHID)          // 6,422,528 halfs = 12.85 MB*2
#define WR_OFF_BYTES (YN_HALFS * 2)             // 16B-aligned offset for wr

// ---------------------------------------------------------------------------
// Kernel 0: reorder + convert conv weights to f16 fragment-friendly layout:
//   wr[tap][cc][n][ci_l]  (tap=kh*3+kw, cc=ci chunk of 32, ci_l=0..31)
// so a lane's B fragment is two contiguous 16-byte loads.
// ---------------------------------------------------------------------------
__global__ __launch_bounds__(256) void weight_reorder(
    const float* __restrict__ w, _Float16* __restrict__ wr)
{
  const int t = blockIdx.x * 256 + threadIdx.x;   // < 9*8*64*32 = 147456
  if (t >= 9 * 8 * 64 * 32) return;
  const int ci_l = t & 31;
  const int n    = (t >> 5) & 63;
  const int cc   = (t >> 11) & 7;
  const int tap  = t >> 14;                       // 0..8
  const int ci   = cc * 32 + ci_l;
  wr[t] = (_Float16)w[(size_t)n * (DIN * 9) + (size_t)ci * 9 + tap];
}

// ---------------------------------------------------------------------------
// Kernel 1: implicit-GEMM conv3x3 s2 p1 (M=100352, N=64, K=2304) with
// v_wmma_f32_16x16x32_f16, fused BN(eval)+ReLU and per-pixel L2 norm.
// Block = 128 threads = 4 waves sharing one 16-pixel M-tile; wave w owns
// channels [16w, 16w+16). A chunk staged once per block through LDS
// (branch-free masked loads), B fragments loaded directly from wr.
// ---------------------------------------------------------------------------
__global__ __launch_bounds__(128) void conv_bn_l2_wmma(
    const float* __restrict__ x, const _Float16* __restrict__ wr,
    const float* __restrict__ gamma, const float* __restrict__ beta,
    const float* __restrict__ mean, const float* __restrict__ var,
    _Float16* __restrict__ yn)
{
  __shared__ __align__(16) _Float16 la[16][32];   // [pixel][ci_local]
  __shared__ float ssq[16];

  const int tid   = threadIdx.x;
  const int lane  = tid & 31;
  const int waveN = tid >> 5;              // 0..3 : 16-channel tile
  const int l16   = lane & 15;
  const int hi    = (lane >> 4) & 1;
  const int baseK = hi ? 8 : 0;
  const int n     = waveN * 16 + l16;      // output channel (B column)

  // cooperative-loader pixel (one of the 16 tile pixels)
  const int p    = tid & 15;
  const int cg   = tid >> 4;               // 0..7 : 4-ci group
  const int mp   = blockIdx.x * 16 + p;
  const int btp  = mp / (HO * WO);
  const int remp = mp % (HO * WO);
  const int ohp  = remp / WO, owp = remp % WO;
  const size_t xbt = (size_t)btp * DIN * HIN * WINP;

  const float sc = gamma[n] * rsqrtf(var[n] + 1e-5f);
  const float mu = mean[n];
  const float bb = beta[n];

  v8f acc = {};
  for (int tap = 0; tap < 9; ++tap) {
    const int kh = tap / 3, kw = tap % 3;
    const int ih = ohp * 2 - 1 + kh;
    const int iw = owp * 2 - 1 + kw;
    const bool valid = (ih >= 0) && (ih < HIN) && (iw >= 0) && (iw < WINP);
    const float maskf = valid ? 1.0f : 0.0f;
    const size_t xoff = xbt + (size_t)(valid ? ih : 0) * WINP + (valid ? iw : 0);

    for (int cc = 0; cc < 8; ++cc) {
      // branch-free cooperative A-chunk load: 16 pixels x 32 ci -> LDS f16
      v4h pk;
#pragma unroll
      for (int j = 0; j < 4; ++j) {
        const int ci = cc * 32 + cg * 4 + j;
        pk[j] = (_Float16)(x[xoff + (size_t)ci * (HIN * WINP)] * maskf);
      }
      *(v4h*)&la[p][cg * 4] = pk;
      __syncthreads();

      // A fragment from LDS (two b128 reads), B fragment from wr (two b128)
      const _Float16* ar = &la[l16][0];
      const v8h alo = *(const v8h*)(ar + baseK);
      const v8h ahi = *(const v8h*)(ar + baseK + 16);
      const v16h a = __builtin_shufflevector(alo, ahi,
          0,1,2,3,4,5,6,7,8,9,10,11,12,13,14,15);
      const _Float16* bp = wr + (((size_t)tap * 8 + cc) * 64 + n) * 32;
      const v8h blo = *(const v8h*)(bp + baseK);
      const v8h bhi = *(const v8h*)(bp + baseK + 16);
      const v16h b = __builtin_shufflevector(blo, bhi,
          0,1,2,3,4,5,6,7,8,9,10,11,12,13,14,15);
      acc = __builtin_amdgcn_wmma_f32_16x16x32_f16(
                false, a, false, b, (short)0, acc, false, false);
      __syncthreads();
    }
  }

  // BN + ReLU
  float v[8];
#pragma unroll
  for (int r = 0; r < 8; ++r) {
    const float t = (acc[r] - mu) * sc + bb;
    v[r] = t > 0.0f ? t : 0.0f;
  }

  // per-pixel sum of squares over all 64 channels via LDS float atomics
  if (tid < 16) ssq[tid] = 0.0f;
  __syncthreads();
#pragma unroll
  for (int r = 0; r < 8; ++r) {
    atomicAdd(&ssq[r + hi * 8], v[r] * v[r]);
  }
  __syncthreads();

#pragma unroll
  for (int r = 0; r < 8; ++r) {
    const int M = r + hi * 8;
    const float inv = rsqrtf(ssq[M] + 1e-6f);
    const int mr = blockIdx.x * 16 + M;
    yn[(size_t)mr * DHID + n] = (_Float16)(v[r] * inv);
  }
}

// ---------------------------------------------------------------------------
// Kernel 2: windowed correlation as a banded GEMM.
// For fixed (b,t,l,u,row h, 16-wide w-tile):
//   C(16x32) = A(16 pixels x 64 ch) * B(64 ch x 32 shifted columns)
// keep the band v = N - M in [0,8]. OOB frames/rows/cols load from clamped
// in-bounds addresses and are zeroed with selects (no EXEC branching).
// ---------------------------------------------------------------------------
__global__ __launch_bounds__(256) void corr_band_wmma(
    const _Float16* __restrict__ yn, float* __restrict__ out)
{
  const int lane = threadIdx.x & 31;
  const int wave = threadIdx.x >> 5;
  int job = blockIdx.x * 8 + wave;          // < 32*5*9*56*4 = 322560
  const int wt = job & 3;  job >>= 2;
  const int h  = job % 56; job /= 56;
  const int u  = job % 9;  job /= 9;
  const int l  = job % 5;  job /= 5;
  const int t  = job % 8;
  const int bi = job / 8;
  const int btIdx = bi * 8 + t;

  const int l16   = lane & 15;
  const int hi    = (lane >> 4) & 1;
  const int baseK = hi ? 8 : 0;

  const int frame = t + l - 2;
  const int hrow  = h + u - 4;
  const bool fval = (frame >= 0) && (frame < 8) && (hrow >= 0) && (hrow < 56);

  const int  w_m  = wt * 16 + l16;
  const bool aval = (w_m < 56);
  const int  w_cl = aval ? w_m : 55;
  const _Float16* arow = yn + ((size_t)(btIdx * 56 + h) * 56 + w_cl) * DHID;
  const _Float16* brow = yn + ((size_t)((bi * 8 + (fval ? frame : 0)) * 56
                                        + (fval ? hrow : 0)) * 56) * DHID;

  v8f acc0 = {}, acc1 = {};
  const v8h zz = {};
#pragma unroll
  for (int s = 0; s < 2; ++s) {
    const int c0 = s * 32 + baseK;
    v8h alo = *(const v8h*)(arow + c0);
    v8h ahi = *(const v8h*)(arow + c0 + 16);
    if (!aval) { alo = zz; ahi = zz; }
    const v16h a = __builtin_shufflevector(alo, ahi,
        0,1,2,3,4,5,6,7,8,9,10,11,12,13,14,15);
#pragma unroll
    for (int nt = 0; nt < 2; ++nt) {
      const int  jcol = wt * 16 - 4 + nt * 16 + l16;
      const bool bv   = fval && (jcol >= 0) && (jcol < 56);
      const int  jcl  = (jcol < 0) ? 0 : (jcol > 55 ? 55 : jcol);
      const _Float16* bp = brow + (size_t)jcl * DHID + c0;
      v8h blo = *(const v8h*)(bp);
      v8h bhi = *(const v8h*)(bp + 16);
      if (!bv) { blo = zz; bhi = zz; }
      const v16h b = __builtin_shufflevector(blo, bhi,
          0,1,2,3,4,5,6,7,8,9,10,11,12,13,14,15);
      if (nt == 0)
        acc0 = __builtin_amdgcn_wmma_f32_16x16x32_f16(
                   false, a, false, b, (short)0, acc0, false, false);
      else
        acc1 = __builtin_amdgcn_wmma_f32_16x16x32_f16(
                   false, a, false, b, (short)0, acc1, false, false);
    }
  }

  // Band store: lane holds C[M, N], M = r + 8*hi, N = nt*16 + l16.
  // out[b,t,h,w,1,l,u,v] with w = wt*16 + M, v = N - M in [0,8].
#pragma unroll
  for (int nt = 0; nt < 2; ++nt) {
    const v8f acc = nt ? acc1 : acc0;
    const int N = nt * 16 + l16;
#pragma unroll
    for (int r = 0; r < 8; ++r) {
      const int M = r + hi * 8;
      const int v = N - M;
      const int w = wt * 16 + M;
      if (v >= 0 && v <= 8 && w < 56) {
        const size_t idx =
            ((((((size_t)btIdx * 56 + h) * 56 + w) * 5 + l) * 9 + u) * 9 + v);
        out[idx] = acc[r];
      }
    }
  }
}

// ---------------------------------------------------------------------------
extern "C" void kernel_launch(void* const* d_in, const int* in_sizes, int n_in,
                              void* d_out, int out_size, void* d_ws, size_t ws_size,
                              hipStream_t stream) {
  const float* x      = (const float*)d_in[0];
  const float* conv_w = (const float*)d_in[1];
  const float* gamma  = (const float*)d_in[2];
  const float* beta   = (const float*)d_in[3];
  const float* mean   = (const float*)d_in[4];
  const float* var    = (const float*)d_in[5];
  float*       out    = (float*)d_out;

  _Float16* yn = (_Float16*)d_ws;                               // 12.85 MB
  _Float16* wr = (_Float16*)((char*)d_ws + WR_OFF_BYTES);       // 0.3 MB

  // Stage 0: weight reorder/convert (9*8*64*32 = 147456 elems)
  weight_reorder<<<(147456 + 255) / 256, 256, 0, stream>>>(conv_w, wr);
  // Stage 1: conv + BN + ReLU + L2 norm -> f16 NHWC features
  conv_bn_l2_wmma<<<NPIX / 16, 128, 0, stream>>>(x, wr, gamma, beta,
                                                 mean, var, yn);
  // Stage 2: banded-GEMM correlation -> fp32 output (covers every element)
  const int jobs = BT * 5 * 9 * 56 * 4;    // 322560, 8 waves per block
  corr_band_wmma<<<jobs / 8, 256, 0, stream>>>(yn, out);
}
